// HeteroGAT_28149215658617
// MI455X (gfx1250) — matrix operand order repaired and checked by reading, not compiled, not run on hardware
//
#include <hip/hip_runtime.h>

#define NNODES 50000
#define NF 128
#define NEDGE 500000
#define NG 64
#define NOUT 10
#define NEG_SLOPE 0.2f

typedef _Float16 v16h __attribute__((ext_vector_type(16)));
typedef _Float16 v8h  __attribute__((ext_vector_type(8)));
typedef float    v8f  __attribute__((ext_vector_type(8)));

// ---------------- utility kernels ----------------

__global__ void k_fill_u32(unsigned* __restrict__ p, unsigned v, int n) {
    int i = blockIdx.x * 256 + threadIdx.x;
    if (i < n) p[i] = v;
}

__global__ void k_f32_to_f16(const float* __restrict__ in, _Float16* __restrict__ out, int n) {
    int i = blockIdx.x * 256 + threadIdx.x;
    if (i < n) out[i] = (_Float16)in[i];
}

// w: [2,4,128(k),128(h)] f32  ->  wt: [2,4,128(h),128(k)] f16 (transposed for WMMA B loads)
__global__ void k_wconv(const float* __restrict__ w, _Float16* __restrict__ wt, int n) {
    int i = blockIdx.x * 256 + threadIdx.x;
    if (i >= n) return;
    int h  = i & 127;
    int k  = (i >> 7) & 127;
    int le = i >> 14;
    wt[(le << 14) + (h << 7) + k] = (_Float16)w[i];
}

// vfold[0..127] = W_src @ a_src ; vfold[128..255] = W_dst @ a_dst
__global__ void k_fold(const float* __restrict__ ws, const float* __restrict__ as,
                       const float* __restrict__ wd, const float* __restrict__ ad,
                       float* __restrict__ vfold) {
    int k = threadIdx.x;
    const float* w = blockIdx.x ? wd : ws;
    const float* a = blockIdx.x ? ad : as;
    float s = 0.f;
    for (int h = 0; h < NF; ++h) s += w[k * NF + h] * a[h];
    vfold[blockIdx.x * NF + k] = s;
}

// s[i] = dot(x[i,:], v) ; one wave per row
__global__ void k_rowdot(const float* __restrict__ x, const float* __restrict__ v,
                         float* __restrict__ s, int nrows) {
    int wid  = (blockIdx.x * blockDim.x + threadIdx.x) >> 5;
    int lane = threadIdx.x & 31;
    if (wid >= nrows) return;
    const float* row = x + (size_t)wid * NF;
    float p = 0.f;
#pragma unroll
    for (int j = 0; j < 4; ++j) { int c = lane + 32 * j; p += row[c] * v[c]; }
    for (int off = 16; off; off >>= 1) p += __shfl_xor(p, off, 32);
    if (lane == 0) s[wid] = p;
}

// ---------------- WMMA GEMM: Y[N x 128] = X[N x 128] (f16) @ W[128 x 128] ----------------
// WT is weight transposed: WT[n][k], f16.  One wave per 16x16 output tile.
__global__ void k_gemm_wmma(const _Float16* __restrict__ X, const _Float16* __restrict__ WT,
                            float* __restrict__ Y) {
    int lane = threadIdx.x & 31;
    int wv   = threadIdx.x >> 5;       // 0..7 : column tile
    int m0   = blockIdx.x * 16;        // row tile base
    int n0   = wv * 16;
    int r    = lane & 15;
    int hi   = lane >> 4;              // 0 or 1 (lane half)
    const _Float16* xrow = X + (size_t)(m0 + r) * NF;
    const _Float16* wrow = WT + (size_t)(n0 + r) * NF;
    v8f c = {};
#pragma unroll
    for (int kt = 0; kt < 4; ++kt) {
        int ko = kt * 32;
        union { v16h v; v8h h[2]; } a, b;
        // A 16x32 f16 layout: lanes0-15 K=0..7 / 16..23 ; lanes16-31 K=8..15 / 24..31
        a.h[0] = *(const v8h*)(xrow + ko + hi * 8);
        a.h[1] = *(const v8h*)(xrow + ko + 16 + hi * 8);
        // B 32x16 f16 layout: lanes0-15 K=0..7 / 16..23 ; lanes16-31 K=8..15 / 24..31 (col per lane)
        b.h[0] = *(const v8h*)(wrow + ko + hi * 8);
        b.h[1] = *(const v8h*)(wrow + ko + 16 + hi * 8);
        c = __builtin_amdgcn_wmma_f32_16x16x32_f16(false, a.v, false, b.v,
                                                   (short)0, c, false, false);
    }
    // C/D layout: VGPR v, lane l -> M = v + 8*(l>>4), N = l&15
    float* yrow = Y + (size_t)(m0 + hi * 8) * NF + n0 + r;
#pragma unroll
    for (int v = 0; v < 8; ++v) yrow[(size_t)v * NF] = c[v];
}

// ---------------- edge kernels ----------------

__device__ __forceinline__ unsigned f32_ord(float f) {
    unsigned u = __float_as_uint(f);
    return (u & 0x80000000u) ? ~u : (u | 0x80000000u);
}
__device__ __forceinline__ float ord_f32(unsigned u) {
    return (u & 0x80000000u) ? __uint_as_float(u ^ 0x80000000u) : __uint_as_float(~u);
}

__global__ void k_edge_score(const int* __restrict__ src, const int* __restrict__ dst,
                             const float* __restrict__ ssrc, const float* __restrict__ sdst,
                             float* __restrict__ score, unsigned* __restrict__ smax) {
    int e = blockIdx.x * 256 + threadIdx.x;
    if (e >= NEDGE) return;
    float v = ssrc[src[e]] + sdst[dst[e]];
    v = (v >= 0.f) ? v : NEG_SLOPE * v;   // leaky_relu
    score[e] = v;
    atomicMax(&smax[dst[e]], f32_ord(v));
}

__global__ void k_edge_expsum(const int* __restrict__ dst, float* __restrict__ score,
                              const unsigned* __restrict__ smax, float* __restrict__ denom,
                              float* __restrict__ cnt) {
    int e = blockIdx.x * 256 + threadIdx.x;
    if (e >= NEDGE) return;
    int d = dst[e];
    float ex = expf(score[e] - ord_f32(smax[d]));
    score[e] = ex;                         // reuse buffer for exp values
    atomicAdd(&denom[d], ex);
    atomicAdd(&cnt[d], 1.0f);
}

// per-node scale = 1/(softmax denom * in-degree) -> folds segment-mean into alpha
__global__ void k_node_wscale(const float* __restrict__ denom, const float* __restrict__ cnt,
                              float* __restrict__ wsc, int n) {
    int j = blockIdx.x * 256 + threadIdx.x;
    if (j >= n) return;
    float d = denom[j];
    wsc[j] = (d > 0.f) ? 1.0f / (d * fmaxf(cnt[j], 1.0f)) : 0.f;
}

// one wave per edge, 4 features per lane
__global__ void k_edge_message(const int* __restrict__ src, const int* __restrict__ dst,
                               const float* __restrict__ ex, const float* __restrict__ wsc,
                               const float* __restrict__ h, float* __restrict__ acc) {
    int t = blockIdx.x * 256 + threadIdx.x;
    int e = t >> 5, lane = t & 31;
    if (e >= NEDGE) return;
    int s = src[e], d = dst[e];
    float coef = ex[e] * wsc[d];
    const float4 hv = *(const float4*)(h + (size_t)s * NF + lane * 4);
    float* ap = acc + (size_t)d * NF + lane * 4;
    atomicAdd(ap + 0, coef * hv.x);
    atomicAdd(ap + 1, coef * hv.y);
    atomicAdd(ap + 2, coef * hv.z);
    atomicAdd(ap + 3, coef * hv.w);
}

// x = relu( (acc + b1 + b2) / 2 )  -- inter-type mean over the 2 edge types per target
__global__ void k_finalize(float* __restrict__ x, const float* __restrict__ acc,
                           const float* __restrict__ b1, const float* __restrict__ b2, int n) {
    int i = blockIdx.x * 256 + threadIdx.x;
    if (i >= n) return;
    int f = i & 127;
    float v = 0.5f * (acc[i] + b1[f] + b2[f]);
    x[i] = (v > 0.f) ? v : 0.f;
}

// ---------------- pooling + classifier ----------------

__global__ void k_pool(const float* __restrict__ x, const int* __restrict__ batch,
                       float* __restrict__ pooled, float* __restrict__ cg, int colbase) {
    int t = blockIdx.x * 256 + threadIdx.x;
    int nid = t >> 5, lane = t & 31;
    if (nid >= NNODES) return;
    int g = batch[nid];
    const float4 v = *(const float4*)(x + (size_t)nid * NF + lane * 4);
    float* p = pooled + (size_t)g * 256 + colbase + lane * 4;
    atomicAdd(p + 0, v.x); atomicAdd(p + 1, v.y);
    atomicAdd(p + 2, v.z); atomicAdd(p + 3, v.w);
    if (lane == 0) atomicAdd(&cg[g], 1.0f);
}

__global__ void k_classify(const float* __restrict__ pooled, const float* __restrict__ cg,
                           const float* __restrict__ wc, const float* __restrict__ bc,
                           float* __restrict__ out) {
    int t = blockIdx.x * blockDim.x + threadIdx.x;
    if (t >= NG * NOUT) return;
    int g = t / NOUT, o = t % NOUT;
    float ia = 1.0f / fmaxf(cg[g], 1.0f);
    float ib = 1.0f / fmaxf(cg[NG + g], 1.0f);
    float s = bc[o];
    for (int k = 0; k < 256; ++k) {
        float v = pooled[g * 256 + k] * (k < 128 ? ia : ib);
        s += v * wc[k * NOUT + o];
    }
    out[t] = s;
}

// ---------------- host orchestration ----------------

extern "C" void kernel_launch(void* const* d_in, const int* in_sizes, int n_in,
                              void* d_out, int out_size, void* d_ws, size_t ws_size,
                              hipStream_t stream) {
    const float* xA_in   = (const float*)d_in[0];
    const float* xB_in   = (const float*)d_in[1];
    const int*   ei[4]   = {(const int*)d_in[2], (const int*)d_in[3],
                            (const int*)d_in[4], (const int*)d_in[5]};
    const int*   batchA  = (const int*)d_in[6];
    const int*   batchB  = (const int*)d_in[7];
    const float* w_src   = (const float*)d_in[9];
    const float* w_dst   = (const float*)d_in[10];
    const float* att_src = (const float*)d_in[11];
    const float* att_dst = (const float*)d_in[12];
    const float* cbias   = (const float*)d_in[13];
    const float* wc      = (const float*)d_in[14];
    const float* bc      = (const float*)d_in[15];
    float* out = (float*)d_out;

    // carve scratch
    char* cur = (char*)d_ws;
    auto carve = [&](size_t bytes) -> char* {
        char* p = cur;
        cur += (bytes + 255) & ~(size_t)255;
        return p;
    };
    const size_t NXF = (size_t)NNODES * NF;                  // 6.4M elems
    float*    xbufA  = (float*)carve(NXF * 4);
    float*    xbufB  = (float*)carve(NXF * 4);
    _Float16* xhA    = (_Float16*)carve(NXF * 2);
    _Float16* xhB    = (_Float16*)carve(NXF * 2);
    _Float16* wtsrc  = (_Float16*)carve((size_t)2 * 4 * 128 * 128 * 2);
    float*    hsrc   = (float*)carve(NXF * 4);
    float*    accA   = (float*)carve(NXF * 4);
    float*    accB   = (float*)carve(NXF * 4);
    float*    ssrc   = (float*)carve((size_t)NNODES * 4);
    float*    sdst   = (float*)carve((size_t)NNODES * 4);
    unsigned* smax   = (unsigned*)carve((size_t)NNODES * 4);
    float*    denom  = (float*)carve((size_t)NNODES * 4);
    float*    cnt    = (float*)carve((size_t)NNODES * 4);
    float*    wsc    = (float*)carve((size_t)NNODES * 4);
    float*    escore = (float*)carve((size_t)NEDGE * 4);
    float*    vfold  = (float*)carve(256 * 4);
    float*    pooled = (float*)carve((size_t)NG * 256 * 4);
    float*    cntg   = (float*)carve((size_t)2 * NG * 4);

    const int WN = 2 * 4 * 128 * 128;  // 131072 weight elems per tensor
    k_wconv<<<(WN + 255) / 256, 256, 0, stream>>>(w_src, wtsrc, WN);
    // fold W_dst@a_dst per edge type below (no full h_dst GEMM needed)

    const int nxf = (int)NXF;
    const int GB_NODE = (NNODES + 255) / 256;
    const int GB_EDGE = (NEDGE + 255) / 256;
    const int st_tab[4] = {0, 0, 1, 1};
    const int dt_tab[4] = {0, 1, 0, 1};

    const float* xcur[2] = {xA_in, xB_in};
    float* xnext[2] = {xbufA, xbufB};

    for (int L = 0; L < 2; ++L) {
        k_f32_to_f16<<<(nxf + 255) / 256, 256, 0, stream>>>(xcur[0], xhA, nxf);
        k_f32_to_f16<<<(nxf + 255) / 256, 256, 0, stream>>>(xcur[1], xhB, nxf);
        k_fill_u32<<<(nxf + 255) / 256, 256, 0, stream>>>((unsigned*)accA, 0u, nxf);
        k_fill_u32<<<(nxf + 255) / 256, 256, 0, stream>>>((unsigned*)accB, 0u, nxf);

        for (int e = 0; e < 4; ++e) {
            int st = st_tab[e], dt = dt_tab[e];
            size_t le   = (size_t)(L * 4 + e);
            size_t woff = le * 128 * 128;
            size_t aoff = le * 128;

            k_fold<<<2, 128, 0, stream>>>(w_src + woff, att_src + aoff,
                                          w_dst + woff, att_dst + aoff, vfold);
            k_rowdot<<<NNODES / 8, 256, 0, stream>>>(xcur[st], vfold, ssrc, NNODES);
            k_rowdot<<<NNODES / 8, 256, 0, stream>>>(xcur[dt], vfold + 128, sdst, NNODES);

            k_gemm_wmma<<<NNODES / 16, 256, 0, stream>>>(st ? xhB : xhA,
                                                         wtsrc + woff, hsrc);

            k_fill_u32<<<GB_NODE, 256, 0, stream>>>(smax, 0u, NNODES);
            k_fill_u32<<<GB_NODE, 256, 0, stream>>>((unsigned*)denom, 0u, NNODES);
            k_fill_u32<<<GB_NODE, 256, 0, stream>>>((unsigned*)cnt, 0u, NNODES);

            const int* esrc = ei[e];
            const int* edst = ei[e] + NEDGE;
            k_edge_score<<<GB_EDGE, 256, 0, stream>>>(esrc, edst, ssrc, sdst, escore, smax);
            k_edge_expsum<<<GB_EDGE, 256, 0, stream>>>(edst, escore, smax, denom, cnt);
            k_node_wscale<<<GB_NODE, 256, 0, stream>>>(denom, cnt, wsc, NNODES);
            k_edge_message<<<NEDGE / 8, 256, 0, stream>>>(esrc, edst, escore, wsc, hsrc,
                                                         dt ? accB : accA);
        }
        // target type 0 receives edge types 0 and 2; type 1 receives 1 and 3
        k_finalize<<<(nxf + 255) / 256, 256, 0, stream>>>(
            xnext[0], accA, cbias + ((size_t)L * 4 + 0) * 128, cbias + ((size_t)L * 4 + 2) * 128, nxf);
        k_finalize<<<(nxf + 255) / 256, 256, 0, stream>>>(
            xnext[1], accB, cbias + ((size_t)L * 4 + 1) * 128, cbias + ((size_t)L * 4 + 3) * 128, nxf);
        xcur[0] = xnext[0];
        xcur[1] = xnext[1];
    }

    k_fill_u32<<<(NG * 256 + 255) / 256, 256, 0, stream>>>((unsigned*)pooled, 0u, NG * 256);
    k_fill_u32<<<1, 256, 0, stream>>>((unsigned*)cntg, 0u, 2 * NG);
    k_pool<<<NNODES / 8, 256, 0, stream>>>(xcur[0], batchA, pooled, cntg, 0);
    k_pool<<<NNODES / 8, 256, 0, stream>>>(xcur[1], batchB, pooled, cntg + NG, 128);
    k_classify<<<(NG * NOUT + 255) / 256, 256, 0, stream>>>(pooled, cntg, wc, bc, out);
}
